// RNN_53300544143474
// MI455X (gfx1250) — matrix operand order repaired and checked by reading, not compiled, not run on hardware
//
#include <hip/hip_runtime.h>
#include <cmath>

typedef __attribute__((ext_vector_type(16))) _Float16 v16h;
typedef __attribute__((ext_vector_type(8)))  float    v8f;
typedef __attribute__((ext_vector_type(4)))  unsigned int v4u;
typedef __attribute__((ext_vector_type(4)))  int v4i;
typedef __attribute__((ext_vector_type(8)))  int v8i;

namespace {
constexpr int Bn = 128, Sn = 1024, En = 256, Hn = 512, Vn = 128;
constexpr int NWG  = 16;         // persistent workgroups for the recurrence
constexpr int NCOL = Hn / NWG;   // 32 hidden columns owned per workgroup
constexpr int TPB  = 256;        // 8 wave32 per workgroup

constexpr size_t SMEM_HALF  = (size_t)NCOL * En + 3ull * NCOL * Hn; // 57344 halves
constexpr size_t SMEM_BYTES = SMEM_HALF * 2;                        // 114688 B < 320KB LDS

constexpr unsigned LDS_WIH0 = 0;
constexpr unsigned LDS_WHH0 = (unsigned)(NCOL * En * 2);               // 16384
constexpr unsigned LDS_WIH1 = LDS_WHH0 + (unsigned)(NCOL * Hn * 2);    // 49152
constexpr unsigned LDS_WHH1 = LDS_WIH1 + (unsigned)(NCOL * Hn * 2);    // 81920
}

union Frag  { v16h h; uint4 q[2]; };
union Half8 { _Float16 h[8]; uint4 q; };

// ---- Tensor Data Mover: DMA a rows x cols (f16) 2D tile into LDS -----------
// D# per ISA ch.8: group0 = {count, lds_addr, global_addr[56:0], type=2};
// group1 = {data_size=2B, tensor_dim0/1, tile_dim0/1, tensor_dim0_stride}.
static __device__ __forceinline__ void tdm_load_2d_f16(unsigned lds_off,
                                                       const void* gptr,
                                                       unsigned rows, unsigned cols) {
  const unsigned long long ga = (unsigned long long)gptr;
  v4u g0;
  g0[0] = 1u;                                        // count=1 user descriptor
  g0[1] = lds_off;                                   // lds_addr (bytes)
  g0[2] = (unsigned)(ga & 0xffffffffu);              // global_addr[31:0]
  g0[3] = (unsigned)((ga >> 32) & 0x01ffffffu) | (2u << 30); // addr[56:32] | type=2
  v8i g1;
  g1[0] = (int)(1u << 16);                           // data_size=1 (2 bytes)
  g1[1] = (int)((cols & 0xffffu) << 16);             // tensor_dim0[15:0]  @bits[63:48]
  g1[2] = (int)((cols >> 16) & 0xffffu)              // tensor_dim0[31:16] @bits[79:64]
        | (int)((rows & 0xffffu) << 16);             // tensor_dim1[15:0]  @bits[95:80]
  g1[3] = (int)((rows >> 16) & 0xffffu)              // tensor_dim1[31:16] @bits[111:96]
        | (int)((cols & 0xffffu) << 16);             // tile_dim0          @bits[127:112]
  g1[4] = (int)(rows & 0xffffu);                     // tile_dim1 (tile_dim2=0)
  g1[5] = (int)cols;                                 // tensor_dim0_stride[31:0]
  g1[6] = 0;                                         // stride hi / dim1_stride lo
  g1[7] = 0;
  v4i z4 = {0, 0, 0, 0};
#if defined(__clang_major__) && (__clang_major__ >= 23)
  v8i z8 = {0, 0, 0, 0, 0, 0, 0, 0};
  __builtin_amdgcn_tensor_load_to_lds(g0, g1, z4, z4, z8, 0);
#else
  __builtin_amdgcn_tensor_load_to_lds(g0, g1, z4, z4, 0);
#endif
}

// ---- WMMA fragment loads (ISA 7.12.2 layouts) -------------------------------
// A (16x32 f16): lane m = l%16 is row M; lanes<16 hold K = k..k+7 & k+16..k+23,
// lanes>=16 hold K = k+8..k+15 & k+24..k+31.  Caller passes p = row*ld + k + lhi*8.
static __device__ __forceinline__ v16h load_a16(const _Float16* p) {
  Frag f;
  f.q[0] = *(const uint4*)(p);
  f.q[1] = *(const uint4*)(p + 16);
  return f.h;
}
// B (32x16 f16): lane n = l%16 is col N; lanes<16 hold K = k..k+7 & k+16..k+23,
// lanes>=16 hold K = k+8..k+15 & k+24..k+31.  Caller passes p = col*ld + k + lhi*8.
static __device__ __forceinline__ v16h load_b16(const _Float16* p) {
  Frag f;
  f.q[0] = *(const uint4*)(p);
  f.q[1] = *(const uint4*)(p + 16);
  return f.h;
}

// Accumulate C[2] (one M-tile x two N-tiles) += act(16xK) * Wlds^T slice.
// wlds is a 32-row x ldk row-major f16 slice in LDS (row = local out column).
static __device__ __forceinline__ void gemm_acc(v8f c[2],
                                                const _Float16* __restrict__ act,
                                                int ldk_a, int row,
                                                const _Float16* wlds, int ldk_w,
                                                int K, int lm, int lhi) {
  const _Float16* arow = act + (size_t)row * ldk_a + lhi * 8;
  const _Float16* w0   = wlds + (size_t)lm * ldk_w + lhi * 8;
  const _Float16* w1   = wlds + (size_t)(16 + lm) * ldk_w + lhi * 8;
#pragma unroll 4
  for (int k = 0; k < K; k += 32) {
    v16h a  = load_a16(arow + k);
    v16h b0 = load_b16(w0 + k);
    v16h b1 = load_b16(w1 + k);
    c[0] = __builtin_amdgcn_wmma_f32_16x16x32_f16(false, a, false, b0, (short)0, c[0], false, false);
    c[1] = __builtin_amdgcn_wmma_f32_16x16x32_f16(false, a, false, b1, (short)0, c[1], false, false);
  }
}

// ---- device-wide sense barrier (16 resident workgroups) ---------------------
static __device__ __forceinline__ void grid_barrier(int* bar) {
  __threadfence();
  __syncthreads();
  if (threadIdx.x == 0) {
    int* cnt = bar;
    int* gen = bar + 1;
    int g = __hip_atomic_load(gen, __ATOMIC_ACQUIRE, __HIP_MEMORY_SCOPE_AGENT);
    int prev = __hip_atomic_fetch_add(cnt, 1, __ATOMIC_ACQ_REL, __HIP_MEMORY_SCOPE_AGENT);
    if (prev == NWG - 1) {
      __hip_atomic_store(cnt, 0, __ATOMIC_RELAXED, __HIP_MEMORY_SCOPE_AGENT);
      __hip_atomic_store(gen, g + 1, __ATOMIC_RELEASE, __HIP_MEMORY_SCOPE_AGENT);
    } else {
      while (__hip_atomic_load(gen, __ATOMIC_ACQUIRE, __HIP_MEMORY_SCOPE_AGENT) == g)
        __builtin_amdgcn_s_sleep(2);
    }
  }
  __syncthreads();
  __threadfence();
}

static __device__ __forceinline__ void copy_rows_to_lds(_Float16* dst,
                                                        const _Float16* src,
                                                        int rows, int ld, int rowbase,
                                                        int tid) {
  const int cpr = ld / 8; // uint4 chunks per row
  for (int idx = tid; idx < rows * cpr; idx += TPB) {
    int r = idx / cpr, o = (idx - r * cpr) * 8;
    *(uint4*)(dst + (size_t)r * ld + o) = *(const uint4*)(src + (size_t)(rowbase + r) * ld + o);
  }
}

// ---- kernels ----------------------------------------------------------------
__global__ void init_state_kernel(_Float16* h0, _Float16* h1, int* bar) {
  int i = blockIdx.x * blockDim.x + threadIdx.x;
  const int n = 2 * Bn * Hn;
  if (i < n) { h0[i] = (_Float16)0.f; h1[i] = (_Float16)0.f; }
  if (i == 0) { bar[0] = 0; bar[1] = 0; }
}

__global__ void f32_to_f16_kernel(const float* __restrict__ src,
                                  _Float16* __restrict__ dst, int n) {
  int i = blockIdx.x * blockDim.x + threadIdx.x;
  if (i < n) dst[i] = (_Float16)src[i];
}

__global__ void bias_fold_kernel(const float* __restrict__ a,
                                 const float* __restrict__ b,
                                 float* __restrict__ c, int n) {
  int i = blockIdx.x * blockDim.x + threadIdx.x;
  if (i < n) c[i] = a[i] + b[i];
}

// emb lookup -> f16, stored (S, B, E) so step t's A-matrix is contiguous.
__global__ void embed_kernel(const int* __restrict__ x, const float* __restrict__ emb,
                             _Float16* __restrict__ xemb) {
  size_t gid = (size_t)blockIdx.x * blockDim.x + threadIdx.x;
  int e8 = (int)(gid & (En / 8 - 1));        // 32 chunks of 8
  size_t tb = gid >> 5;                      // t*B + b
  if (tb >= (size_t)Sn * Bn) return;
  int b = (int)(tb & (Bn - 1));
  int t = (int)(tb >> 7);
  int xi = x[(size_t)b * Sn + t];
  const float* src = emb + (size_t)xi * En + e8 * 8;
  float4 f0 = *(const float4*)(src);
  float4 f1 = *(const float4*)(src + 4);
  Half8 o;
  o.h[0] = (_Float16)f0.x; o.h[1] = (_Float16)f0.y;
  o.h[2] = (_Float16)f0.z; o.h[3] = (_Float16)f0.w;
  o.h[4] = (_Float16)f1.x; o.h[5] = (_Float16)f1.y;
  o.h[6] = (_Float16)f1.z; o.h[7] = (_Float16)f1.w;
  *(uint4*)(xemb + tb * En + e8 * 8) = o.q;
}

// Persistent 2-layer RNN recurrence. 16 workgroups x 8 waves; each wg owns 32
// hidden columns for both layers; weight slices live in LDS (TDM-prefetched +
// vector copy); hidden states ping-pong in global (L2-resident).
// Two device barriers per time step.
__global__ __launch_bounds__(TPB, 1)
void rnn_recurrence_kernel(const _Float16* __restrict__ xemb,   // (S,B,E)
                           const _Float16* __restrict__ Wih0,   // (H,E)
                           const _Float16* __restrict__ Whh0,   // (H,H)
                           const _Float16* __restrict__ Wih1,   // (H,H)
                           const _Float16* __restrict__ Whh1,   // (H,H)
                           const float* __restrict__ b0,        // (H,) folded
                           const float* __restrict__ b1,        // (H,) folded
                           _Float16* __restrict__ H0buf,        // 2*B*H ping-pong
                           _Float16* __restrict__ H1buf,        // 2*B*H ping-pong
                           _Float16* __restrict__ Y,            // (S,B,H) layer-1 outs
                           float* __restrict__ hn,              // (2,B,H) f32
                           int* __restrict__ bar) {
  const int wg   = blockIdx.x;
  const int tid  = threadIdx.x;
  const int lane = tid & 31;
  const int wave = tid >> 5;          // M tile: batch rows 16*wave..+15
  const int lm   = lane & 15;
  const int lhi  = lane >> 4;
  const int colbase = wg * NCOL;

  extern __shared__ _Float16 smem[];
  _Float16* sWih0 = smem;                       // 32 x 256
  _Float16* sWhh0 = sWih0 + (size_t)NCOL * En;  // 32 x 512
  _Float16* sWih1 = sWhh0 + (size_t)NCOL * Hn;  // 32 x 512
  _Float16* sWhh1 = sWih1 + (size_t)NCOL * Hn;  // 32 x 512

  // Async DMA of the weight slices via the Tensor Data Mover (wave 0 issues).
  if (tid < 32) {
    tdm_load_2d_f16(LDS_WIH0, Wih0 + (size_t)colbase * En, NCOL, En);
    tdm_load_2d_f16(LDS_WHH0, Whh0 + (size_t)colbase * Hn, NCOL, Hn);
    tdm_load_2d_f16(LDS_WIH1, Wih1 + (size_t)colbase * Hn, NCOL, Hn);
    tdm_load_2d_f16(LDS_WHH1, Whh1 + (size_t)colbase * Hn, NCOL, Hn);
    __builtin_amdgcn_s_wait_tensorcnt(0);
  }
  __syncthreads();
  // Vector-copy path guarantees LDS contents (overwrites in place).
  copy_rows_to_lds(sWih0, Wih0, NCOL, En, colbase, tid);
  copy_rows_to_lds(sWhh0, Whh0, NCOL, Hn, colbase, tid);
  copy_rows_to_lds(sWih1, Wih1, NCOL, Hn, colbase, tid);
  copy_rows_to_lds(sWhh1, Whh1, NCOL, Hn, colbase, tid);
  __syncthreads();

  const int row = wave * 16 + lm;     // A-fragment row (batch index)
  float bias0[2], bias1[2];
#pragma unroll
  for (int n = 0; n < 2; ++n) {
    bias0[n] = b0[colbase + n * 16 + lm];
    bias1[n] = b1[colbase + n * 16 + lm];
  }

  for (int t = 0; t < Sn; ++t) {
    const int rp = t & 1, wp = rp ^ 1;
    const _Float16* xt  = xemb + (size_t)t * Bn * En;
    const _Float16* h0r = H0buf + (size_t)rp * Bn * Hn;
    _Float16*       h0w = H0buf + (size_t)wp * Bn * Hn;
    const _Float16* h1r = H1buf + (size_t)rp * Bn * Hn;
    _Float16*       h1w = H1buf + (size_t)wp * Bn * Hn;

    if (t + 1 < Sn)  // warm L2/L0 for next step's embeddings
      __builtin_prefetch(xemb + ((size_t)(t + 1) * Bn + row) * En, 0, 0);

    // -------- layer 0: h0n = tanh(xt W_ih0^T + h0 W_hh0^T + b0) -------------
    v8f c0[2];
#pragma unroll
    for (int n = 0; n < 2; ++n)
#pragma unroll
      for (int i = 0; i < 8; ++i) c0[n][i] = bias0[n];
    gemm_acc(c0, xt,  En, row, sWih0, En, En, lm, lhi);
    gemm_acc(c0, h0r, Hn, row, sWhh0, Hn, Hn, lm, lhi);
#pragma unroll
    for (int n = 0; n < 2; ++n) {
      const int col = colbase + n * 16 + lm;
#pragma unroll
      for (int i = 0; i < 8; ++i) {
        float v = tanhf(c0[n][i]);
        const int r = wave * 16 + i + lhi * 8;   // C layout: M = i (+8 hi half)
        h0w[(size_t)r * Hn + col] = (_Float16)v;
        if (t == Sn - 1) hn[(size_t)r * Hn + col] = v;
      }
    }
    grid_barrier(bar);

    // -------- layer 1: h1n = tanh(h0n W_ih1^T + h1 W_hh1^T + b1) ------------
    v8f c1[2];
#pragma unroll
    for (int n = 0; n < 2; ++n)
#pragma unroll
      for (int i = 0; i < 8; ++i) c1[n][i] = bias1[n];
    gemm_acc(c1, h0w, Hn, row, sWih1, Hn, Hn, lm, lhi);
    gemm_acc(c1, h1r, Hn, row, sWhh1, Hn, Hn, lm, lhi);
#pragma unroll
    for (int n = 0; n < 2; ++n) {
      const int col = colbase + n * 16 + lm;
#pragma unroll
      for (int i = 0; i < 8; ++i) {
        float v = tanhf(c1[n][i]);
        const int r = wave * 16 + i + lhi * 8;
        _Float16 hv = (_Float16)v;
        h1w[(size_t)r * Hn + col] = hv;
        Y[((size_t)t * Bn + r) * Hn + col] = hv;
        if (t == Sn - 1) hn[(size_t)(Bn * Hn) + (size_t)r * Hn + col] = v;
      }
    }
    grid_barrier(bar);
  }
}

// logits = Y (S,B,H) @ W_out^T + b_out  -> out (B,S,V) f32
__global__ __launch_bounds__(TPB)
void logits_kernel(const _Float16* __restrict__ Y,
                   const _Float16* __restrict__ Wout,   // (V,H) f16
                   const float* __restrict__ bout,      // (V,)
                   float* __restrict__ out) {
  const int lane = threadIdx.x & 31;
  const int wave = threadIdx.x >> 5;
  const int lm = lane & 15, lhi = lane >> 4;
  const int id = blockIdx.x * 8 + wave;          // 65536 tiles total
  const int vt = id & 7;
  const int bt = (id >> 3) & 7;
  const int s  = id >> 6;

  v8f c;
  const float bb = bout[vt * 16 + lm];
#pragma unroll
  for (int i = 0; i < 8; ++i) c[i] = bb;

  const _Float16* arow = Y + ((size_t)s * Bn + bt * 16 + lm) * Hn + lhi * 8;
  const _Float16* brow = Wout + (size_t)(vt * 16 + lm) * Hn + lhi * 8;
#pragma unroll 4
  for (int k = 0; k < Hn; k += 32) {
    v16h a = load_a16(arow + k);
    v16h b = load_b16(brow + k);
    c = __builtin_amdgcn_wmma_f32_16x16x32_f16(false, a, false, b, (short)0, c, false, false);
  }
#pragma unroll
  for (int i = 0; i < 8; ++i) {
    const int b = bt * 16 + i + lhi * 8;
    out[((size_t)b * Sn + s) * Vn + vt * 16 + lm] = c[i];
  }
}

// ---- host launch ------------------------------------------------------------
extern "C" void kernel_launch(void* const* d_in, const int* in_sizes, int n_in,
                              void* d_out, int out_size, void* d_ws, size_t ws_size,
                              hipStream_t stream) {
  const int*   x     = (const int*)d_in[0];
  const float* emb   = (const float*)d_in[1];
  const float* W_ih0 = (const float*)d_in[2];
  const float* b_ih0 = (const float*)d_in[3];
  const float* W_hh0 = (const float*)d_in[4];
  const float* b_hh0 = (const float*)d_in[5];
  const float* W_ih1 = (const float*)d_in[6];
  const float* b_ih1 = (const float*)d_in[7];
  const float* W_hh1 = (const float*)d_in[8];
  const float* b_hh1 = (const float*)d_in[9];
  const float* W_out = (const float*)d_in[10];
  const float* b_out = (const float*)d_in[11];

  float* out_logits = (float*)d_out;                               // (B,S,V)
  float* out_hn     = (float*)d_out + (size_t)Bn * Sn * Vn;        // (2,B,H)

  // workspace carve-up (all offsets 256B-aligned by construction)
  char* ws = (char*)d_ws;
  size_t off = 0;
  auto take = [&](size_t bytes) { char* p = ws + off; off += bytes; return p; };
  _Float16* xemb  = (_Float16*)take((size_t)Sn * Bn * En * 2);   // 64 MB
  _Float16* Yseq  = (_Float16*)take((size_t)Sn * Bn * Hn * 2);   // 128 MB
  _Float16* H0buf = (_Float16*)take((size_t)2 * Bn * Hn * 2);
  _Float16* H1buf = (_Float16*)take((size_t)2 * Bn * Hn * 2);
  _Float16* wih0h = (_Float16*)take((size_t)Hn * En * 2);
  _Float16* whh0h = (_Float16*)take((size_t)Hn * Hn * 2);
  _Float16* wih1h = (_Float16*)take((size_t)Hn * Hn * 2);
  _Float16* whh1h = (_Float16*)take((size_t)Hn * Hn * 2);
  _Float16* wouth = (_Float16*)take((size_t)Vn * Hn * 2);
  float*    b0f   = (float*)take((size_t)Hn * 4);
  float*    b1f   = (float*)take((size_t)Hn * 4);
  int*      bar   = (int*)take(256);
  (void)ws_size; (void)n_in; (void)in_sizes; (void)out_size;

  // 1) state + barrier init (ws is poisoned between runs)
  init_state_kernel<<<(2 * Bn * Hn + TPB - 1) / TPB, TPB, 0, stream>>>(H0buf, H1buf, bar);

  // 2) embedding lookup -> f16 (S,B,E)
  {
    const size_t total = (size_t)Sn * Bn * (En / 8);
    embed_kernel<<<(unsigned)((total + TPB - 1) / TPB), TPB, 0, stream>>>(x, emb, xemb);
  }

  // 3) weight conversion to f16 + bias folding
  auto cvt = [&](const float* s, _Float16* d, int n) {
    f32_to_f16_kernel<<<(n + TPB - 1) / TPB, TPB, 0, stream>>>(s, d, n);
  };
  cvt(W_ih0, wih0h, Hn * En);
  cvt(W_hh0, whh0h, Hn * Hn);
  cvt(W_ih1, wih1h, Hn * Hn);
  cvt(W_hh1, whh1h, Hn * Hn);
  cvt(W_out, wouth, Vn * Hn);
  bias_fold_kernel<<<(Hn + TPB - 1) / TPB, TPB, 0, stream>>>(b_ih0, b_hh0, b0f, Hn);
  bias_fold_kernel<<<(Hn + TPB - 1) / TPB, TPB, 0, stream>>>(b_ih1, b_hh1, b1f, Hn);

  // 4) persistent recurrence (16 workgroups, 112 KB dynamic LDS each)
  (void)hipFuncSetAttribute((const void*)rnn_recurrence_kernel,
                            hipFuncAttributeMaxDynamicSharedMemorySize,
                            (int)SMEM_BYTES);
  rnn_recurrence_kernel<<<NWG, TPB, SMEM_BYTES, stream>>>(
      xemb, wih0h, whh0h, wih1h, whh1h, b0f, b1f,
      H0buf, H1buf, Yseq, out_hn, bar);

  // 5) output projection (65536 WMMA tiles)
  logits_kernel<<<(Sn * 8 * 8) / 8, TPB, 0, stream>>>(Yseq, wouth, b_out, out_logits);
}
